// FLAAttentionWrapper_66795331388110
// MI455X (gfx1250) — matrix operand order repaired and checked by reading, not compiled, not run on hardware
//
#include <hip/hip_runtime.h>
#include <hip/hip_bf16.h>

// ---------------------------------------------------------------------------
// Types for CDNA5 WMMA (wave32): V_WMMA_F32_16X16X32_BF16
// ---------------------------------------------------------------------------
typedef __bf16 bf16_t;
typedef __attribute__((ext_vector_type(16))) __bf16 v16bf;
typedef __attribute__((ext_vector_type(8)))  float  v8f;

__device__ __forceinline__ float  bf2f(bf16_t x) { return (float)x; }
__device__ __forceinline__ bf16_t f2bf(float x)  { return (bf16_t)x; }

__device__ __forceinline__ v8f wmma_bf16(v16bf a, v16bf b, v8f c) {
  // 8 args: (neg_a, A, neg_b, B, c_mod, C, reuse_a, reuse_b)
  return __builtin_amdgcn_wmma_f32_16x16x32_bf16(false, a, false, b,
                                                 (short)0, c, false, false);
}

// ---------------------------------------------------------------------------
// Tensor Data Mover: 2D tile Global->LDS with LDS padding (ISA ch.8 D#)
//   group0: count=1 | lds_addr | global_addr(57b) | type=2
//   group1: data_size, pad_enable/interval/amount, tensor/tile dims, stride0
//   tensor_dim0/1 fixed at 2^20 (our tiles are always in-bounds -> no OOB)
// This toolchain ships the 6-arg builtin form:
//   (uint32x4 g0, int32x8 g1, int32x4, int32x4, int32x8, i32 cpol)
// ---------------------------------------------------------------------------
#if defined(__has_builtin)
#if __has_builtin(__builtin_amdgcn_tensor_load_to_lds) && __has_builtin(__builtin_amdgcn_s_wait_tensorcnt)
#define USE_TDM 1
#endif
#endif
#ifndef USE_TDM
#define USE_TDM 0
#endif

#if USE_TDM
typedef __attribute__((ext_vector_type(4))) unsigned int u32x4;
typedef __attribute__((ext_vector_type(8))) int          i32x8;
typedef __attribute__((ext_vector_type(4))) int          i32x4;

__device__ __forceinline__ void tdm_load_2d(unsigned int lds_off, const void* gptr,
                                            unsigned int dsz_code,   // 1=2B, 2=4B
                                            unsigned int tile_d0,    // elements
                                            unsigned int tile_d1,    // rows
                                            unsigned int stride0,    // elements
                                            unsigned int pad_i_code, // 2^(c+1) dwords
                                            unsigned int pad_a_code) // c+1 dwords
{
  unsigned long long ga = (unsigned long long)(uintptr_t)gptr;
  u32x4 g0;
  g0[0] = 1u;                                         // count=1, user mode
  g0[1] = lds_off;                                    // LDS byte address
  g0[2] = (unsigned int)ga;                           // global_addr[31:0]
  g0[3] = (unsigned int)((ga >> 32) & 0x01FFFFFFu) | (2u << 30);  // [56:32]|type=2
  i32x8 g1;
  g1[0] = (int)((dsz_code << 16) | (1u << 20) | (pad_i_code << 22) | (pad_a_code << 25));
  g1[1] = 0;                                          // abar=0, tensor_dim0 lo16 = 0
  g1[2] = (int)0x10;                                  // tensor_dim0 hi16 = 0x10 (2^20), dim1 lo16 = 0
  g1[3] = (int)(0x10u | (tile_d0 << 16));             // tensor_dim1 hi16 = 0x10, tile_dim0
  g1[4] = (int)tile_d1;                               // tile_dim1 (tile_dim2 = 0)
  g1[5] = (int)stride0;                               // tensor_dim0_stride[31:0]
  g1[6] = 0;
  g1[7] = 0;
  i32x4 z4 = {0, 0, 0, 0};
  i32x8 z8 = {0, 0, 0, 0, 0, 0, 0, 0};
  __builtin_amdgcn_tensor_load_to_lds(g0, g1, z4, z4, z8, 0);
}
#endif

// A-fragment: 16x32 bf16, lane = row (M):
//   element e holds K = (lane>>4)*16 + e
__device__ __forceinline__ v16bf load_a_frag(const bf16_t* __restrict__ p,
                                             int ld, int lane) {
  const int m = lane & 15, h = lane >> 4;
  const bf16_t* r = p + m * ld + h * 16;
  v16bf a;
#pragma unroll
  for (int q = 0; q < 8; ++q) { a[q]     = r[q]; }
#pragma unroll
  for (int q = 0; q < 8; ++q) { a[8 + q] = r[8 + q]; }
  return a;
}

// B-fragment from an [N][K]-major LDS view (K contiguous per column owner):
//   lane = col (N), element e holds K = (lane>>4)*16 + e
__device__ __forceinline__ v16bf load_b_frag_nk(const bf16_t* __restrict__ p,
                                                int ld, int lane) {
  const int n = lane & 15, h = lane >> 4;
  const bf16_t* c = p + n * ld + h * 16;
  v16bf b;
#pragma unroll
  for (int e = 0; e < 16; ++e) b[e] = c[e];
  return b;
}

// B-fragment from a [K][N]-major LDS view (strided K reads)
__device__ __forceinline__ v16bf load_b_frag_kn(const bf16_t* __restrict__ p,
                                                int ld, int lane) {
  const int n = lane & 15, h = lane >> 4;
  const bf16_t* c = p + (h * 16) * ld + n;
  v16bf b;
#pragma unroll
  for (int e = 0; e < 16; ++e) b[e] = c[e * ld];
  return b;
}

// ---------------------------------------------------------------------------
// Elementwise conversion kernels
// ---------------------------------------------------------------------------
__global__ void cvt_f32_to_bf16(const float* __restrict__ in,
                                bf16_t* __restrict__ out, int n) {
  int i = blockIdx.x * blockDim.x + threadIdx.x;
  if (i < n) out[i] = f2bf(in[i]);
}

// Pack [Wq|Wk|Wv|Wog] (each 1024x1024 row-major) into Wcat 1024x4096 bf16
__global__ void pack_wcat(const float* __restrict__ wq, const float* __restrict__ wk,
                          const float* __restrict__ wv, const float* __restrict__ wog,
                          bf16_t* __restrict__ out) {
  int i = blockIdx.x * 256 + threadIdx.x;        // 0 .. 1024*4096-1
  int k = i >> 12, n = i & 4095;
  int sel = n >> 10, c = n & 1023;
  const float* src = (sel == 0) ? wq : (sel == 1) ? wk : (sel == 2) ? wv : wog;
  out[i] = f2bf(src[k * 1024 + c]);
}

// ---------------------------------------------------------------------------
// Gate kernel: gk = log_sigmoid(x @ Wg1 @ Wg2) / 16  (fp32, [8192,1024])
// ---------------------------------------------------------------------------
__global__ __launch_bounds__(256)
void gate_kernel(const float* __restrict__ x, const float* __restrict__ Wg1,
                 const float* __restrict__ Wg2, float* __restrict__ gk) {
  __shared__ float red[16 * 257];
  __shared__ float tmp[16];
  const int row = blockIdx.x;          // 0..8191
  const int tid = threadIdx.x;
  float p[16];
#pragma unroll
  for (int j = 0; j < 16; ++j) p[j] = 0.f;
  for (int c = tid; c < 1024; c += 256) {
    float xv = x[(size_t)row * 1024 + c];
#pragma unroll
    for (int j = 0; j < 16; ++j) p[j] += xv * Wg1[c * 16 + j];
  }
#pragma unroll
  for (int j = 0; j < 16; ++j) red[j * 257 + tid] = p[j];
  __syncthreads();
  if (tid < 16) {
    float s = 0.f;
    for (int t = 0; t < 256; ++t) s += red[tid * 257 + t];
    tmp[tid] = s;
  }
  __syncthreads();
  for (int c = tid; c < 1024; c += 256) {
    float z = 0.f;
#pragma unroll
    for (int j = 0; j < 16; ++j) z += tmp[j] * Wg2[j * 1024 + c];
    float ls = fminf(z, 0.f) - log1pf(expf(-fabsf(z)));   // log_sigmoid(z)
    gk[(size_t)row * 1024 + c] = ls * (1.0f / 16.0f);
  }
}

// ---------------------------------------------------------------------------
// Generic bf16 WMMA GEMM: C[MxN] = A[MxK] @ B[KxN]
//   block tile 128x128, 8 waves of 64x32, K-step 32
//   A tile staged by TDM (tensor_load_to_lds, padded rows), B tile staged
//   transposed by VALU/DS. First scale_ncols columns scaled by `scale`.
// ---------------------------------------------------------------------------
#define GT_M 128
#define GT_N 128
#define GT_K 32
#define A_LD 36   // 32 + 4 pad (bf16)  == 16 dwords + 2 dwords TDM pad
#define B_LD 40   // 32 + 8 pad (bf16)

template <bool OUT_BF16>
__global__ __launch_bounds__(256)
void gemm_bf16(const bf16_t* __restrict__ A, const bf16_t* __restrict__ B,
               void* __restrict__ Cout, int M, int N, int K,
               float scale, int scale_ncols) {
  __shared__ bf16_t sA[GT_M * A_LD];
  __shared__ bf16_t sBT[GT_N * B_LD];   // transposed: [n][k]
  const int tid = threadIdx.x, lane = tid & 31, wv = tid >> 5;
  const int m0 = blockIdx.y * GT_M, n0 = blockIdx.x * GT_N;
  const int wm = (wv >> 2) * 64;        // 2 waves along M
  const int wn = (wv & 3) * 32;         // 4 waves along N

  v8f acc[4][2];
#pragma unroll
  for (int i = 0; i < 4; ++i)
#pragma unroll
    for (int j = 0; j < 2; ++j)
#pragma unroll
      for (int r = 0; r < 8; ++r) acc[i][j][r] = 0.f;

  const int b_k = tid >> 3, b_nc = (tid & 7) * 16;

  for (int kt = 0; kt < K; kt += GT_K) {
    // ---- stage A tile 128x32 via Tensor Data Mover (pad -> stride A_LD) ----
#if USE_TDM
    if (wv == 0) {
      tdm_load_2d((unsigned int)(uintptr_t)(void*)sA,
                  A + (size_t)m0 * K + kt,
                  /*dsz=2B*/1, /*tile_d0*/GT_K, /*tile_d1*/GT_M,
                  /*stride0*/(unsigned int)K,
                  /*pad_interval: 16 dwords*/3, /*pad_amount: 2 dwords*/1);
    }
#else
    {
      const int a_row = tid >> 1, a_c16 = (tid & 1) * 16;
      const uint4* g = reinterpret_cast<const uint4*>(
          A + (size_t)(m0 + a_row) * K + kt + a_c16);
      uint4 d0 = g[0], d1 = g[1];
      uint32_t* dst = reinterpret_cast<uint32_t*>(sA + a_row * A_LD + a_c16);
      dst[0] = d0.x; dst[1] = d0.y; dst[2] = d0.z; dst[3] = d0.w;
      dst[4] = d1.x; dst[5] = d1.y; dst[6] = d1.z; dst[7] = d1.w;
    }
#endif
    // ---- stage B tile 32x128, stored transposed [n][k] ----
    {
      union { uint4 u; bf16_t h[8]; } u0, u1;
      const uint4* g = reinterpret_cast<const uint4*>(
          B + (size_t)(kt + b_k) * N + n0 + b_nc);
      u0.u = g[0]; u1.u = g[1];
#pragma unroll
      for (int e = 0; e < 8; ++e) sBT[(b_nc + e) * B_LD + b_k] = u0.h[e];
#pragma unroll
      for (int e = 0; e < 8; ++e) sBT[(b_nc + 8 + e) * B_LD + b_k] = u1.h[e];
      if (kt + GT_K < K)   // prefetch next B tile -> global_prefetch_b8
        __builtin_prefetch(B + (size_t)(kt + GT_K + b_k) * N + n0 + b_nc, 0, 1);
    }
#if USE_TDM
    if (wv == 0) __builtin_amdgcn_s_wait_tensorcnt(0);
#endif
    __syncthreads();
    // ---- WMMA: 4 M-tiles x 2 N-tiles per wave ----
    v16bf af[4], bfm[2];
#pragma unroll
    for (int i = 0; i < 4; ++i)
      af[i] = load_a_frag(sA + (wm + 16 * i) * A_LD, A_LD, lane);
#pragma unroll
    for (int j = 0; j < 2; ++j)
      bfm[j] = load_b_frag_nk(sBT + (wn + 16 * j) * B_LD, B_LD, lane);
#pragma unroll
    for (int i = 0; i < 4; ++i)
#pragma unroll
      for (int j = 0; j < 2; ++j)
        acc[i][j] = wmma_bf16(af[i], bfm[j], acc[i][j]);
    __syncthreads();
  }

  // ---- epilogue: C layout lane=(n, m-half), VGPR r -> m = half*8 + r ----
  const int cn = lane & 15, mh = (lane >> 4) * 8;
#pragma unroll
  for (int i = 0; i < 4; ++i)
#pragma unroll
    for (int j = 0; j < 2; ++j) {
      int gcol = n0 + wn + 16 * j + cn;
      float sc = (gcol < scale_ncols) ? scale : 1.0f;
#pragma unroll
      for (int r = 0; r < 8; ++r) {
        int grow = m0 + wm + 16 * i + mh + r;
        float v = acc[i][j][r] * sc;
        if (OUT_BF16)
          ((bf16_t*)Cout)[(size_t)grow * N + gcol] = f2bf(v);
        else
          ((float*)Cout)[(size_t)grow * N + gcol] = v;
      }
    }
}

// ---------------------------------------------------------------------------
// GLA core: one block per (b,h); sequential scan over 32 chunks of 64.
// State S (dk x dv = 128x128 fp32) lives in registers, distributed 8 waves.
// ---------------------------------------------------------------------------
#define CHK 64
#define DKD 128
#define LDB16 136   // 128 + 8 pad, bf16
#define LDKS  72    // 64 + 8 pad, bf16
#define LDAT  72    // 64 + 8 pad, bf16 (A tile rows are 64 wide)
#define LDF   132   // 128 + 4 pad, fp32 == 128 dwords + 4 dwords TDM pad
#define NCHUNK 32

__global__ __launch_bounds__(256)
void gla_kernel(const bf16_t* __restrict__ qkvog, const float* __restrict__ gkbuf,
                const float* __restrict__ gnorm, bf16_t* __restrict__ obuf,
                int T) {
  __shared__ bf16_t sSt[DKD * LDB16];   // S^T: [dv][dk] bf16 mirror
  __shared__ bf16_t sQs[CHK * LDB16];   // q * exp(b)           [t][dk]
  __shared__ bf16_t sKd[CHK * LDB16];   // k * exp(-b)          [t][dk]
  __shared__ bf16_t sKsT[DKD * LDKS];   // (k*exp(blast-b))^T   [dk][t]
  __shared__ bf16_t sV[CHK * LDB16];    // v                    [t][dv]
  __shared__ float  sGO[CHK * LDF];     // gk-cumsum, reused as o tile
  __shared__ bf16_t sA[CHK * LDAT];     // masked intra scores  [i][j]
  __shared__ float  sBlast[DKD];
  __shared__ float  sGn[DKD];
  __shared__ float  sRed[CHK * 4];

  const int tid = threadIdx.x, lane = tid & 31, wv = tid >> 5;
  const int b = blockIdx.x >> 3, h = blockIdx.x & 7;
  const size_t base_q  = (size_t)b * T * 4096 + (size_t)h * 128;
  const size_t base_k  = base_q + 1024;
  const size_t base_v  = base_q + 2048;
  const size_t base_og = base_q + 3072;
  const size_t base_g  = (size_t)b * T * 1024 + (size_t)h * 128;

  // ---- init ----
  for (int i = tid; i < DKD * LDB16; i += 256) sSt[i] = f2bf(0.f);
  if (tid < DKD) sGn[tid] = gnorm[tid];
  v8f Sacc[2][4];
#pragma unroll
  for (int mi = 0; mi < 2; ++mi)
#pragma unroll
    for (int ni = 0; ni < 4; ++ni)
#pragma unroll
      for (int r = 0; r < 8; ++r) Sacc[mi][ni][r] = 0.f;
  const int sm = (wv & 3) * 32;   // this wave's dk rows of S
  const int sn = (wv >> 2) * 64;  // this wave's dv cols of S
  __syncthreads();

  for (int ch = 0; ch < NCHUNK; ++ch) {
    const int t0 = ch * CHK;
    // 1) gk chunk -> sGO (TDM 2D tile: 128 floats x 64 rows, padded to LDF)
#if USE_TDM
    if (wv == 0) {
      tdm_load_2d((unsigned int)(uintptr_t)(void*)sGO,
                  gkbuf + base_g + (size_t)t0 * 1024,
                  /*dsz=4B*/2, /*tile_d0*/DKD, /*tile_d1*/CHK,
                  /*stride0*/1024,
                  /*pad_interval: 128 dwords*/6, /*pad_amount: 4 dwords*/3);
      __builtin_amdgcn_s_wait_tensorcnt(0);
    }
#else
    for (int i = tid; i < CHK * DKD; i += 256) {
      int r = i >> 7, c = i & 127;
      sGO[r * LDF + c] = gkbuf[base_g + (size_t)(t0 + r) * 1024 + c];
    }
#endif
    __syncthreads();
    // 2) in-place column cumsum over chunk dim
    if (tid < DKD) {
      float s = 0.f;
      for (int r = 0; r < CHK; ++r) {
        s += sGO[r * LDF + tid];
        sGO[r * LDF + tid] = s;
      }
      sBlast[tid] = s;
    }
    __syncthreads();
    // 3) load q,k,v and apply exp gating -> bf16 tiles
    for (int i = tid; i < CHK * DKD; i += 256) {
      int r = i >> 7, c = i & 127;
      float bg = sGO[r * LDF + c];
      size_t gr = (size_t)(t0 + r) * 4096;
      float qv = bf2f(qkvog[base_q + gr + c]);
      float kv = bf2f(qkvog[base_k + gr + c]);
      float vv = bf2f(qkvog[base_v + gr + c]);
      float eb = expf(bg);
      sQs[r * LDB16 + c] = f2bf(qv * eb);
      sKd[r * LDB16 + c] = f2bf(kv / eb);
      sKsT[c * LDKS + r] = f2bf(kv * expf(sBlast[c] - bg));
      sV[r * LDB16 + c]  = f2bf(vv);
    }
    __syncthreads();
    // 4) A = q_s @ kd^T (64x64, K=128), causal mask, store bf16
    {
      const int am = (wv & 3) * 16, an = (wv >> 2) * 32;
      v8f a0, a1;
#pragma unroll
      for (int r = 0; r < 8; ++r) { a0[r] = 0.f; a1[r] = 0.f; }
      for (int ks = 0; ks < DKD; ks += 32) {
        v16bf af = load_a_frag(sQs + am * LDB16 + ks, LDB16, lane);
        v16bf b0 = load_b_frag_nk(sKd + an * LDB16 + ks, LDB16, lane);
        v16bf b1 = load_b_frag_nk(sKd + (an + 16) * LDB16 + ks, LDB16, lane);
        a0 = wmma_bf16(af, b0, a0);
        a1 = wmma_bf16(af, b1, a1);
      }
      const int cn = lane & 15, mh = (lane >> 4) * 8;
#pragma unroll
      for (int r = 0; r < 8; ++r) {
        int irow = am + mh + r;
        int j0 = an + cn, j1 = an + 16 + cn;
        sA[irow * LDAT + j0] = f2bf((j0 <= irow) ? a0[r] : 0.f);
        sA[irow * LDAT + j1] = f2bf((j1 <= irow) ? a1[r] : 0.f);
      }
    }
    __syncthreads();
    // 5) o = A @ v  +  q_s @ S^T   (64x128)
    {
      const int om = (wv & 3) * 16, on = (wv >> 2) * 64;
      v8f oc[4];
#pragma unroll
      for (int j = 0; j < 4; ++j)
#pragma unroll
        for (int r = 0; r < 8; ++r) oc[j][r] = 0.f;
      for (int ks = 0; ks < CHK; ks += 32) {              // intra
        v16bf af = load_a_frag(sA + om * LDAT + ks, LDAT, lane);
#pragma unroll
        for (int j = 0; j < 4; ++j) {
          v16bf bb = load_b_frag_kn(sV + ks * LDB16 + on + 16 * j, LDB16, lane);
          oc[j] = wmma_bf16(af, bb, oc[j]);
        }
      }
      for (int ks = 0; ks < DKD; ks += 32) {              // inter
        v16bf af = load_a_frag(sQs + om * LDB16 + ks, LDB16, lane);
#pragma unroll
        for (int j = 0; j < 4; ++j) {
          v16bf bb = load_b_frag_nk(sSt + (on + 16 * j) * LDB16 + ks, LDB16, lane);
          oc[j] = wmma_bf16(af, bb, oc[j]);
        }
      }
      const int cn = lane & 15, mh = (lane >> 4) * 8;
#pragma unroll
      for (int j = 0; j < 4; ++j)
#pragma unroll
        for (int r = 0; r < 8; ++r)
          sGO[(om + mh + r) * LDF + on + 16 * j + cn] = oc[j][r];
    }
    __syncthreads();   // all S^T reads & o writes complete
    // 6) S <- exp(b_last) (.) S + k_s^T @ v ; refresh S^T mirror
    {
      const int cn = lane & 15, mh = (lane >> 4) * 8;
#pragma unroll
      for (int mi = 0; mi < 2; ++mi)
#pragma unroll
        for (int r = 0; r < 8; ++r) {
          float f = expf(sBlast[sm + 16 * mi + mh + r]);
#pragma unroll
          for (int ni = 0; ni < 4; ++ni) Sacc[mi][ni][r] *= f;
        }
      for (int ks = 0; ks < CHK; ks += 32) {
        v16bf a0 = load_a_frag(sKsT + sm * LDKS + ks, LDKS, lane);
        v16bf a1 = load_a_frag(sKsT + (sm + 16) * LDKS + ks, LDKS, lane);
#pragma unroll
        for (int ni = 0; ni < 4; ++ni) {
          v16bf bb = load_b_frag_kn(sV + ks * LDB16 + sn + 16 * ni, LDB16, lane);
          Sacc[0][ni] = wmma_bf16(a0, bb, Sacc[0][ni]);
          Sacc[1][ni] = wmma_bf16(a1, bb, Sacc[1][ni]);
        }
      }
#pragma unroll
      for (int mi = 0; mi < 2; ++mi)
#pragma unroll
        for (int ni = 0; ni < 4; ++ni)
#pragma unroll
          for (int r = 0; r < 8; ++r) {
            int dk_i = sm + 16 * mi + mh + r;
            int dv_i = sn + 16 * ni + cn;
            sSt[dv_i * LDB16 + dk_i] = f2bf(Sacc[mi][ni][r]);
          }
    }
    // 7) RMS-norm + SiLU(og) gate, write bf16 rows
    {
      int r = tid >> 2, qd = tid & 3;
      float ss = 0.f;
      for (int c = qd * 32; c < qd * 32 + 32; ++c) {
        float v = sGO[r * LDF + c];
        ss += v * v;
      }
      sRed[r * 4 + qd] = ss;
    }
    __syncthreads();
    {
      int r = tid >> 2, qd = tid & 3;
      float tot = sRed[r * 4] + sRed[r * 4 + 1] + sRed[r * 4 + 2] + sRed[r * 4 + 3];
      float rn = rsqrtf(tot * (1.0f / 128.0f) + 1e-6f);
      size_t ob  = (size_t)(b * T + t0 + r) * 1024 + (size_t)h * 128;
      size_t ogr = base_og + (size_t)(t0 + r) * 4096;
      for (int c = qd * 32; c < qd * 32 + 32; ++c) {
        float og = bf2f(qkvog[ogr + c]);
        float sw = og / (1.f + expf(-og));        // og * sigmoid(og)
        obuf[ob + c] = f2bf(sGO[r * LDF + c] * rn * sGn[c] * sw);
      }
    }
    __syncthreads();   // sGO / sSt safe to reuse next chunk
  }
}

// ---------------------------------------------------------------------------
// Host launcher
// ---------------------------------------------------------------------------
extern "C" void kernel_launch(void* const* d_in, const int* in_sizes, int n_in,
                              void* d_out, int out_size, void* d_ws, size_t ws_size,
                              hipStream_t stream) {
  (void)in_sizes; (void)n_in; (void)out_size; (void)ws_size;
  const float* x    = (const float*)d_in[0];
  const float* Wq   = (const float*)d_in[1];
  const float* Wk   = (const float*)d_in[2];
  const float* Wv   = (const float*)d_in[3];
  const float* Wg1  = (const float*)d_in[4];
  const float* Wg2  = (const float*)d_in[5];
  const float* Wog  = (const float*)d_in[6];
  const float* gnrm = (const float*)d_in[7];
  const float* Wo   = (const float*)d_in[8];

  const int Bsz = 4, T = 2048, D = 1024;
  const int M = Bsz * T;                 // 8192

  // workspace carve-out (256B aligned)
  char* w = (char*)d_ws;
  auto carve = [&](size_t bytes) {
    char* p = w;
    w += (bytes + 255) & ~(size_t)255;
    return p;
  };
  bf16_t* x_bf   = (bf16_t*)carve((size_t)M * D * 2);        // 16 MB
  bf16_t* wcat   = (bf16_t*)carve((size_t)D * 4 * D * 2);    //  8 MB
  bf16_t* wo_bf  = (bf16_t*)carve((size_t)D * D * 2);        //  2 MB
  bf16_t* qkvog  = (bf16_t*)carve((size_t)M * 4 * D * 2);    // 64 MB
  float*  gkbuf  = (float*)carve((size_t)M * D * 4);         // 32 MB
  bf16_t* obuf   = (bf16_t*)carve((size_t)M * D * 2);        // 16 MB

  // 1) fp32 -> bf16 conversions
  cvt_f32_to_bf16<<<(M * D) / 256, 256, 0, stream>>>(x, x_bf, M * D);
  pack_wcat<<<(D * 4 * D) / 256, 256, 0, stream>>>(Wq, Wk, Wv, Wog, wcat);
  cvt_f32_to_bf16<<<(D * D) / 256, 256, 0, stream>>>(Wo, wo_bf, D * D);

  // 2) gate path (fp32)
  gate_kernel<<<M, 256, 0, stream>>>(x, Wg1, Wg2, gkbuf);

  // 3) fused projections: [Q(scaled)|K|V|OG] = x @ [Wq|Wk|Wv|Wog]
  {
    dim3 grid((4 * D) / GT_N, M / GT_M);   // 32 x 64
    const float qscale = 0.08838834764831845f;   // dk^-0.5
    gemm_bf16<true><<<grid, 256, 0, stream>>>(x_bf, wcat, qkvog,
                                              M, 4 * D, D, qscale, D);
  }

  // 4) chunked GLA scan + norm + SiLU gate
  gla_kernel<<<Bsz * 8, 256, 0, stream>>>(qkvog, gkbuf, gnrm, obuf, T);

  // 5) output projection: d_out = o @ Wo (fp32 out)
  {
    dim3 grid(D / GT_N, M / GT_M);         // 8 x 64
    gemm_bf16<false><<<grid, 256, 0, stream>>>(obuf, wo_bf, d_out,
                                               M, D, D, 1.0f, 0);
  }
}
